// MoE_51797305589879
// MI455X (gfx1250) — matrix operand order repaired and checked by reading, not compiled, not run on hardware
//
#include <hip/hip_runtime.h>
#include <math.h>

#define D_   1024
#define H_   4096
#define E_   8
#define N_   8192   // B*S tokens
#define MT   32     // token tile (2 x 16 WMMA rows)
#define HC   256    // H chunk width (8 waves x 32 cols)

typedef __attribute__((ext_vector_type(16))) _Float16 v16h;
typedef __attribute__((ext_vector_type(8)))  float    v8f;

__device__ __forceinline__ v8f wmma_f16(v16h a, v16h b, v8f c) {
  // D = A(16x32,f16) x B(32x16,f16) + C(16x16,f32)
  return __builtin_amdgcn_wmma_f32_16x16x32_f16(false, a, false, b, (short)0, c,
                                                false, false);
}

// B fragment (32x16 f16) gathered from row-major fp32 W (leading dim ld):
// rows k0..k0+31, cols col0..col0+15.  ISA layout: lanes 0-15 -> K=(j<8?j:8+j),
// lanes 16-31 -> K=8+(j<8?j:8+j), N = lane%16.  (fallback path)
__device__ __forceinline__ v16h load_b_frag_f32(const float* __restrict__ W,
                                                int ld, int k0, int col0) {
  const int lane = threadIdx.x & 31;
  const float* p = W + (size_t)(k0 + ((lane >> 4) << 3)) * ld + (col0 + (lane & 15));
  __builtin_prefetch(p + (size_t)32 * ld, 0, 0);
  v16h b;
#pragma unroll
  for (int j = 0; j < 16; ++j) b[j] = (_Float16)p[(size_t)(j < 8 ? j : j + 8) * ld];
  return b;
}

__device__ __forceinline__ float gelu_exact(float hv) {
  return 0.5f * hv * (1.f + erff(hv * 0.70710678118f));
}

// ---------------- Router: logits -> softmax -> top-2 -> per-expert lists ----
__global__ __launch_bounds__(256) void MoE_router(
    const float* __restrict__ x, const float* __restrict__ Wg,
    const float* __restrict__ bg, int* __restrict__ cnt,
    int* __restrict__ tlist, float* __restrict__ tgate) {
  const int lane = threadIdx.x & 31;
  const int n = blockIdx.x * 8 + (threadIdx.x >> 5);   // one wave per token
  const float* xr = x + (size_t)n * D_;
  float acc[E_];
#pragma unroll
  for (int e = 0; e < E_; ++e) acc[e] = 0.f;
  for (int d = lane; d < D_; d += 32) {
    const float xv = xr[d];
    const float* wr = Wg + d * E_;
#pragma unroll
    for (int e = 0; e < E_; ++e) acc[e] += xv * wr[e];
  }
#pragma unroll
  for (int e = 0; e < E_; ++e) {
#pragma unroll
    for (int off = 16; off > 0; off >>= 1) acc[e] += __shfl_xor(acc[e], off, 32);
  }
  if (lane == 0) {
    float lg[E_], mx = -3.0e38f;
#pragma unroll
    for (int e = 0; e < E_; ++e) { lg[e] = acc[e] + bg[e]; mx = fmaxf(mx, lg[e]); }
    float s = 0.f;
#pragma unroll
    for (int e = 0; e < E_; ++e) { lg[e] = expf(lg[e] - mx); s += lg[e]; }
    const float inv = 1.f / s;
    int i1 = 0; float p1 = -1.f;
#pragma unroll
    for (int e = 0; e < E_; ++e) if (lg[e] > p1) { p1 = lg[e]; i1 = e; }
    int i2 = 0; float p2 = -1.f;
#pragma unroll
    for (int e = 0; e < E_; ++e) if (e != i1 && lg[e] > p2) { p2 = lg[e]; i2 = e; }
    int pos = atomicAdd(&cnt[i1], 1);
    tlist[i1 * N_ + pos] = n;  tgate[i1 * N_ + pos] = p1 * inv;
    pos = atomicAdd(&cnt[i2], 1);
    tlist[i2 * N_ + pos] = n;  tgate[i2 * N_ + pos] = p2 * inv;
  }
}

// -------- prep: fp32 weights -> f16 in B-fragment-tiled layout --------------
// Wf[((e*ncb + cb)*nkb + kb)*512 + lane*16 + j] = W[e][kb*32 + 8*(lane>=16) + (j<8 ? j : 8+j)][cb*16 + lane%16]
// One wave per 512-element fragment block; write = contiguous 32B per lane.
__global__ __launch_bounds__(256) void MoE_cvt_w(const float* __restrict__ W,
                                                 _Float16* __restrict__ Wf,
                                                 int KD, int CD) {
  const int lane = threadIdx.x & 31;
  const int gw = blockIdx.x * 8 + (threadIdx.x >> 5);  // linear frag block
  const int nkb = KD >> 5, ncb = CD >> 4;
  const int kb  = gw % nkb;
  const int cbe = gw / nkb;            // e*ncb + cb
  const int cb  = cbe % ncb;
  const int e   = cbe / ncb;
  const float* p = W + (size_t)e * KD * CD +
                   (size_t)(kb * 32 + ((lane >> 4) << 3)) * CD + cb * 16 + (lane & 15);
  v16h b;
#pragma unroll
  for (int j = 0; j < 16; ++j) b[j] = (_Float16)p[(size_t)(j < 8 ? j : j + 8) * CD];
  *(v16h*)(Wf + (size_t)gw * 512 + lane * 16) = b;
}

// -------- prep: x rows -> f16 pre-swizzled in per-row A-fragment order ------
// xh[n*1024 + kb*32 + hi*16 + j] = x[n][kb*32 + 8*hi + (j<8 ? j : 8+j)]
__global__ __launch_bounds__(256) void MoE_cvt_x(const float* __restrict__ x,
                                                 _Float16* __restrict__ xh) {
  const int t = blockIdx.x * 256 + threadIdx.x;        // N*64 units of 16 halves
  const int n = t >> 6, rem = t & 63;
  const int kb = rem >> 1, hi = rem & 1;
  const float* src = x + (size_t)n * D_ + kb * 32 + hi * 8;
  float4 a0 = *(const float4*)(src);
  float4 a1 = *(const float4*)(src + 4);
  float4 a2 = *(const float4*)(src + 16);
  float4 a3 = *(const float4*)(src + 20);
  v16h o;
  o[0]=(_Float16)a0.x; o[1]=(_Float16)a0.y; o[2]=(_Float16)a0.z; o[3]=(_Float16)a0.w;
  o[4]=(_Float16)a1.x; o[5]=(_Float16)a1.y; o[6]=(_Float16)a1.z; o[7]=(_Float16)a1.w;
  o[8]=(_Float16)a2.x; o[9]=(_Float16)a2.y; o[10]=(_Float16)a2.z; o[11]=(_Float16)a2.w;
  o[12]=(_Float16)a3.x; o[13]=(_Float16)a3.y; o[14]=(_Float16)a3.z; o[15]=(_Float16)a3.w;
  *(v16h*)(xh + (size_t)n * D_ + kb * 32 + hi * 16) = o;
}

// ---------------- Expert FFN, f16 fast path --------------------------------
__global__ __launch_bounds__(256) void MoE_ffn_f16(
    const _Float16* __restrict__ xh,
    const _Float16* __restrict__ W1f, const float* __restrict__ b1,
    const _Float16* __restrict__ W2f, const float* __restrict__ b2,
    const int* __restrict__ cnt, const int* __restrict__ tlist,
    const float* __restrict__ tgate, float* __restrict__ out) {
  const int e  = blockIdx.y;
  const int t0 = blockIdx.x * MT;
  const int ce = cnt[e];
  if (t0 >= ce) return;
  const int nt = min(MT, ce - t0);

  __shared__ __align__(32) _Float16 Xa[2 * 32 * 512];        // X tile 32x1024 (64KB)
  __shared__ __align__(32) _Float16 Ha[2 * (HC / 32) * 512]; // H chunk 32xHC (16KB)
  __shared__ int   tok_l[MT];
  __shared__ float gate_l[MT];

  const int tid  = threadIdx.x;
  const int lane = tid & 31;
  const int w    = tid >> 5;
  const int hi   = lane >> 4;

  // ---- stage X tile: async byte-copy (xh already f16 + A-swizzled per row).
  // Row r occupies 2KB of xh; 8 threads/row x 16 x 16B async-to-LDS each.
  {
    const int row = tid >> 3;
    const int sub = tid & 7;
    const bool valid = row < nt;
    int tok = 0; float g = 0.f;
    if (valid) { tok = tlist[e * N_ + t0 + row]; g = tgate[e * N_ + t0 + row]; }
    if (sub == 0) { tok_l[row] = tok; gate_l[row] = g; }
    const int mb = row >> 4, rl = row & 15;
    const char* src_base = (const char*)xh + (size_t)tok * (D_ * 2);
    const unsigned lds_base = (unsigned)(uintptr_t)&Xa[0];
#pragma unroll
    for (int t = 0; t < 16; ++t) {
      const int v    = sub * 16 + t;        // 16B unit within the row (0..127)
      const int kb   = v >> 2;
      const int hi2  = (v >> 1) & 1;
      const int half = v & 1;
      const int idx  = ((mb * 32 + kb) * 512 + (rl + 16 * hi2) * 16) + half * 8;
      if (valid) {
        const unsigned dst = lds_base + (unsigned)(idx * 2);
        const void* srcp = src_base + (kb * 64 + hi2 * 32 + half * 16);
        asm volatile("global_load_async_to_lds_b128 %0, %1, off"
                     :: "v"(dst), "v"(srcp) : "memory");
      } else {
        *(uint4*)&Xa[idx] = make_uint4(0u, 0u, 0u, 0u);   // zero padding rows
      }
    }
  }
  asm volatile("s_wait_asynccnt 0x0" ::: "memory");
  __syncthreads();

  const _Float16* W1e = W1f + ((size_t)e * (H_ / 16)) * (D_ / 32) * 512;
  const _Float16* W2e = W2f + ((size_t)e * (D_ / 16)) * (H_ / 32) * 512;

  v8f acc2[2][8];
#pragma unroll
  for (int mb = 0; mb < 2; ++mb)
#pragma unroll
    for (int nb = 0; nb < 8; ++nb)
#pragma unroll
      for (int r = 0; r < 8; ++r) acc2[mb][nb][r] = 0.f;

  for (int hc = 0; hc < H_; hc += HC) {
    // ---- phase 1: Hchunk = gelu(X @ W1[:, hc:hc+HC] + b1); wave -> 32 cols
    v8f acc1[2][2];
#pragma unroll
    for (int mb = 0; mb < 2; ++mb)
#pragma unroll
      for (int nb = 0; nb < 2; ++nb)
#pragma unroll
        for (int r = 0; r < 8; ++r) acc1[mb][nb][r] = 0.f;

    const int cb0 = (hc >> 4) + w * 2;            // col block base (W1 tiling)
    for (int k0 = 0; k0 < D_; k0 += 32) {
      const int kb = k0 >> 5;
      v16h a0 = *(const v16h*)&Xa[(0 * 32 + kb) * 512 + lane * 16];
      v16h a1 = *(const v16h*)&Xa[(1 * 32 + kb) * 512 + lane * 16];
#pragma unroll
      for (int nb = 0; nb < 2; ++nb) {
        const _Float16* fp = W1e + ((size_t)(cb0 + nb) * (D_ / 32) + kb) * 512 + lane * 16;
        __builtin_prefetch(fp + 512, 0, 0);       // next K block, contiguous 1KB away
        v16h b = *(const v16h*)fp;
        acc1[0][nb] = wmma_f16(a0, b, acc1[0][nb]);
        acc1[1][nb] = wmma_f16(a1, b, acc1[1][nb]);
      }
    }
    // bias + exact gelu, store f16 chunk to LDS in A-swizzle for phase 2
#pragma unroll
    for (int mb = 0; mb < 2; ++mb)
#pragma unroll
      for (int nb = 0; nb < 2; ++nb) {
        const int kc = w * 32 + nb * 16 + (lane & 15);
        const float bv = b1[(size_t)e * H_ + hc + kc];
        const int l2h = 16 * ((kc >> 3) & 1);
        const int j2  = (kc & 7) + ((kc & 16) ? 8 : 0);
        const int kb2 = kc >> 5;
#pragma unroll
        for (int r = 0; r < 8; ++r) {
          const float hv = gelu_exact(acc1[mb][nb][r] + bv);
          const int rw = (hi << 3) + r;
          Ha[(mb * (HC / 32) + kb2) * 512 + (rw + l2h) * 16 + j2] = (_Float16)hv;
        }
      }
    __syncthreads();

    // ---- phase 2: acc2 += Hchunk @ W2[hc:hc+HC, :]; wave -> cols w*128..+127
    for (int k0 = 0; k0 < HC; k0 += 32) {
      const int kb = k0 >> 5;
      const int kbg = (hc + k0) >> 5;
      v16h a0 = *(const v16h*)&Ha[(0 * (HC / 32) + kb) * 512 + lane * 16];
      v16h a1 = *(const v16h*)&Ha[(1 * (HC / 32) + kb) * 512 + lane * 16];
#pragma unroll
      for (int nb = 0; nb < 8; ++nb) {
        const _Float16* fp = W2e + ((size_t)(w * 8 + nb) * (H_ / 32) + kbg) * 512 + lane * 16;
        __builtin_prefetch(fp + 512, 0, 0);
        v16h b = *(const v16h*)fp;
        acc2[0][nb] = wmma_f16(a0, b, acc2[0][nb]);
        acc2[1][nb] = wmma_f16(a1, b, acc2[1][nb]);
      }
    }
    __syncthreads();   // before next chunk overwrites Ha
  }

  // ---- epilogue: +b2, gate, accumulate into zeroed output (K=2 adders) ----
#pragma unroll
  for (int mb = 0; mb < 2; ++mb)
#pragma unroll
    for (int nb = 0; nb < 8; ++nb) {
      const int col = w * 128 + nb * 16 + (lane & 15);
      const float b2v = b2[(size_t)e * D_ + col];
#pragma unroll
      for (int r = 0; r < 8; ++r) {
        const int row = (mb << 4) + (hi << 3) + r;
        if (row < nt) {
          const float v = (acc2[mb][nb][r] + b2v) * gate_l[row];
          __hip_atomic_fetch_add(&out[(size_t)tok_l[row] * D_ + col], v,
                                 __ATOMIC_RELAXED, __HIP_MEMORY_SCOPE_AGENT);
        }
      }
    }
}

// ---------------- Expert FFN, fp32-direct fallback (small workspace) -------
__global__ __launch_bounds__(256) void MoE_ffn(
    const float* __restrict__ x,
    const float* __restrict__ W1, const float* __restrict__ b1,
    const float* __restrict__ W2, const float* __restrict__ b2,
    const int* __restrict__ cnt, const int* __restrict__ tlist,
    const float* __restrict__ tgate, float* __restrict__ out) {
  const int e  = blockIdx.y;
  const int t0 = blockIdx.x * MT;
  const int ce = cnt[e];
  if (t0 >= ce) return;
  const int nt = min(MT, ce - t0);

  __shared__ __align__(32) _Float16 Xa[2 * 32 * 512];
  __shared__ __align__(32) _Float16 Ha[2 * (HC / 32) * 512];
  __shared__ int   tok_l[MT];
  __shared__ float gate_l[MT];

  const int tid  = threadIdx.x;
  const int lane = tid & 31;
  const int w    = tid >> 5;
  const int hi   = lane >> 4;

  {
    const int row = tid >> 3;
    const int sub = tid & 7;
    int tok = 0; float g = 0.f;
    const bool valid = row < nt;
    if (valid) { tok = tlist[e * N_ + t0 + row]; g = tgate[e * N_ + t0 + row]; }
    if (sub == 0) { tok_l[row] = tok; gate_l[row] = g; }
    const float4* xr = (const float4*)(x + (size_t)tok * D_) + sub * 32;
    const int mb = row >> 4;
    const int rl = row & 15;
    for (int i = 0; i < 32; ++i) {
      float4 v = valid ? xr[i] : make_float4(0.f, 0.f, 0.f, 0.f);
      const int d0 = sub * 128 + i * 4;
      const int kb = d0 >> 5;
      const int l2 = rl + (((d0 >> 3) & 1) << 4);
      const int j2 = (d0 & 7) + ((d0 & 16) ? 8 : 0);
      _Float16* dst = &Xa[((mb << 5) + kb) * 512 + l2 * 16 + j2];
      dst[0] = (_Float16)v.x; dst[1] = (_Float16)v.y;
      dst[2] = (_Float16)v.z; dst[3] = (_Float16)v.w;
    }
  }
  __syncthreads();

  const float* W1e = W1 + (size_t)e * D_ * H_;
  const float* W2e = W2 + (size_t)e * H_ * D_;

  v8f acc2[2][8];
#pragma unroll
  for (int mb = 0; mb < 2; ++mb)
#pragma unroll
    for (int nb = 0; nb < 8; ++nb)
#pragma unroll
      for (int r = 0; r < 8; ++r) acc2[mb][nb][r] = 0.f;

  for (int hc = 0; hc < H_; hc += HC) {
    v8f acc1[2][2];
#pragma unroll
    for (int mb = 0; mb < 2; ++mb)
#pragma unroll
      for (int nb = 0; nb < 2; ++nb)
#pragma unroll
        for (int r = 0; r < 8; ++r) acc1[mb][nb][r] = 0.f;

    const int cw = w * 32;
    for (int k0 = 0; k0 < D_; k0 += 32) {
      const int kb = k0 >> 5;
      v16h a0 = *(const v16h*)&Xa[(0 * 32 + kb) * 512 + lane * 16];
      v16h a1 = *(const v16h*)&Xa[(1 * 32 + kb) * 512 + lane * 16];
#pragma unroll
      for (int nb = 0; nb < 2; ++nb) {
        v16h b = load_b_frag_f32(W1e, H_, k0, hc + cw + nb * 16);
        acc1[0][nb] = wmma_f16(a0, b, acc1[0][nb]);
        acc1[1][nb] = wmma_f16(a1, b, acc1[1][nb]);
      }
    }
#pragma unroll
    for (int mb = 0; mb < 2; ++mb)
#pragma unroll
      for (int nb = 0; nb < 2; ++nb) {
        const int kc = cw + nb * 16 + (lane & 15);
        const float bv = b1[(size_t)e * H_ + hc + kc];
        const int l2h = 16 * ((kc >> 3) & 1);
        const int j2  = (kc & 7) + ((kc & 16) ? 8 : 0);
        const int kb2 = kc >> 5;
#pragma unroll
        for (int r = 0; r < 8; ++r) {
          const float hv = gelu_exact(acc1[mb][nb][r] + bv);
          const int rw = (hi << 3) + r;
          Ha[(mb * (HC / 32) + kb2) * 512 + (rw + l2h) * 16 + j2] = (_Float16)hv;
        }
      }
    __syncthreads();

    for (int k0 = 0; k0 < HC; k0 += 32) {
      const int kb = k0 >> 5;
      v16h a0 = *(const v16h*)&Ha[(0 * (HC / 32) + kb) * 512 + lane * 16];
      v16h a1 = *(const v16h*)&Ha[(1 * (HC / 32) + kb) * 512 + lane * 16];
#pragma unroll
      for (int nb = 0; nb < 8; ++nb) {
        v16h b = load_b_frag_f32(W2e, D_, hc + k0, w * 128 + nb * 16);
        acc2[0][nb] = wmma_f16(a0, b, acc2[0][nb]);
        acc2[1][nb] = wmma_f16(a1, b, acc2[1][nb]);
      }
    }
    __syncthreads();
  }

#pragma unroll
  for (int mb = 0; mb < 2; ++mb)
#pragma unroll
    for (int nb = 0; nb < 8; ++nb) {
      const int col = w * 128 + nb * 16 + (lane & 15);
      const float b2v = b2[(size_t)e * D_ + col];
#pragma unroll
      for (int r = 0; r < 8; ++r) {
        const int row = (mb << 4) + (hi << 3) + r;
        if (row < nt) {
          const float v = (acc2[mb][nb][r] + b2v) * gate_l[row];
          __hip_atomic_fetch_add(&out[(size_t)tok_l[row] * D_ + col], v,
                                 __ATOMIC_RELAXED, __HIP_MEMORY_SCOPE_AGENT);
        }
      }
    }
}

extern "C" void kernel_launch(void* const* d_in, const int* in_sizes, int n_in,
                              void* d_out, int out_size, void* d_ws, size_t ws_size,
                              hipStream_t stream) {
  (void)in_sizes; (void)n_in;
  const float* x  = (const float*)d_in[0];
  const float* Wg = (const float*)d_in[1];
  const float* bg = (const float*)d_in[2];
  const float* W1 = (const float*)d_in[3];
  const float* b1 = (const float*)d_in[4];
  const float* W2 = (const float*)d_in[5];
  const float* b2 = (const float*)d_in[6];
  float* out = (float*)d_out;

  // workspace layout (1KB-aligned regions)
  const size_t off_tlist = 1024;
  const size_t off_tgate = off_tlist + (size_t)E_ * N_ * sizeof(int);   // 263168
  const size_t off_xh    = 1u << 20;                                    // 1 MiB
  const size_t off_w1f   = off_xh  + (size_t)N_ * D_ * 2;               // +16MiB
  const size_t off_w2f   = off_w1f + (size_t)E_ * D_ * H_ * 2;          // +64MiB
  const size_t need_f16  = off_w2f + (size_t)E_ * H_ * D_ * 2;          // ~145MiB

  char* ws = (char*)d_ws;
  int*      cnt   = (int*)ws;
  int*      tlist = (int*)(ws + off_tlist);
  float*    tgate = (float*)(ws + off_tgate);

  hipMemsetAsync(cnt, 0, 1024, stream);
  hipMemsetAsync(d_out, 0, (size_t)out_size * sizeof(float), stream);

  MoE_router<<<N_ / 8, 256, 0, stream>>>(x, Wg, bg, cnt, tlist, tgate);

  dim3 grid(N_ / MT, E_);   // covers worst case (all tokens on one expert)

  if (ws_size >= need_f16) {
    _Float16* xh  = (_Float16*)(ws + off_xh);
    _Float16* W1f = (_Float16*)(ws + off_w1f);
    _Float16* W2f = (_Float16*)(ws + off_w2f);
    MoE_cvt_x<<<(N_ * 64) / 256, 256, 0, stream>>>(x, xh);
    MoE_cvt_w<<<(E_ * (H_ / 16) * (D_ / 32)) / 8, 256, 0, stream>>>(W1, W1f, D_, H_);
    MoE_cvt_w<<<(E_ * (D_ / 16) * (H_ / 32)) / 8, 256, 0, stream>>>(W2, W2f, H_, D_);
    MoE_ffn_f16<<<grid, 256, 0, stream>>>(xh, W1f, b1, W2f, b2,
                                          cnt, tlist, tgate, out);
  } else {
    MoE_ffn<<<grid, 256, 0, stream>>>(x, W1, b1, W2, b2, cnt, tlist, tgate, out);
  }
}
